// PointNetPlusSAModule_7576322310711
// MI455X (gfx1250) — matrix operand order repaired and checked by reading, not compiled, not run on hardware
//
#include <hip/hip_runtime.h>

// =====================================================================
// PointNet++ SA module for MI455X (gfx1250, wave32).
// GEMM layers use v_wmma_f32_16x16x32_f16 (f16 in, f32 accum),
// templated on (COUT, CINP): branch-free fully-unrolled WMMA inner loop
// with a lean store-only epilogue. BN stats computed by a separate
// bandwidth-bound reduction kernel (deterministic two-stage, no atomics).
// Workspace (~476 MB): samp, new_pc, gidx, Wh(f16), partials, st,
//                      bufA (96ch f16), bufB (128ch f16).
// =====================================================================

typedef __attribute__((ext_vector_type(16))) _Float16 v16h;
typedef __attribute__((ext_vector_type(8)))  _Float16 v8h;
typedef __attribute__((ext_vector_type(8)))  float    v8f;

#define B_ 16
#define N_ 4096
#define M_ 1024
#define GX_ 64    // col-group grid dim for GEMM
#define SGX_ 128  // fixed block count for stats reduction (partials sizing)

// ----------------------------------------------------------------------
// Farthest point sampling: 1 block per batch, points cached in LDS.
// samp[b][it] follows the reference scan: idx0 = 0, then argmax(dmin).
// ----------------------------------------------------------------------
__global__ __launch_bounds__(1024) void fps_kernel(const float* __restrict__ pc,
                                                   int* __restrict__ samp) {
  __shared__ float px[N_], py[N_], pz[N_];
  __shared__ float rv[32];
  __shared__ int   ri[32];
  __shared__ int   bestShared;
  const int b = blockIdx.x, tid = threadIdx.x, lane = tid & 31, wid = tid >> 5;
  const float* p = pc + (size_t)b * 3 * N_;
  for (int j = tid; j < N_; j += 1024) {
    px[j] = p[j]; py[j] = p[N_ + j]; pz[j] = p[2 * N_ + j];
  }
  float dmin[4];
#pragma unroll
  for (int t = 0; t < 4; ++t) dmin[t] = 1e10f;
  int far = 0;
  __syncthreads();
  for (int it = 0; it < M_; ++it) {
    if (tid == 0) samp[b * M_ + it] = far;
    const float cx = px[far], cy = py[far], cz = pz[far];
    float bv = -1e30f; int bi = N_;
#pragma unroll
    for (int t = 0; t < 4; ++t) {
      const int j = tid + t * 1024;
      const float dx = px[j] - cx, dy = py[j] - cy, dz = pz[j] - cz;
      const float d = dx * dx + dy * dy + dz * dz;
      dmin[t] = fminf(dmin[t], d);
      if (dmin[t] > bv || (dmin[t] == bv && j < bi)) { bv = dmin[t]; bi = j; }
    }
#pragma unroll
    for (int off = 16; off; off >>= 1) {
      const float ov = __shfl_xor(bv, off);
      const int   oi = __shfl_xor(bi, off);
      if (ov > bv || (ov == bv && oi < bi)) { bv = ov; bi = oi; }
    }
    if (lane == 0) { rv[wid] = bv; ri[wid] = bi; }
    __syncthreads();
    if (wid == 0) {
      bv = rv[lane]; bi = ri[lane];
#pragma unroll
      for (int off = 16; off; off >>= 1) {
        const float ov = __shfl_xor(bv, off);
        const int   oi = __shfl_xor(bi, off);
        if (ov > bv || (ov == bv && oi < bi)) { bv = ov; bi = oi; }
      }
      if (lane == 0) bestShared = bi;
    }
    __syncthreads();
    far = bestShared;
  }
}

// ----------------------------------------------------------------------
// Gather sampled centers: new_pc[b][d][m] -> ws copy + d_out head.
// ----------------------------------------------------------------------
__global__ void gather_newpc(const float* __restrict__ pc, const int* __restrict__ samp,
                             float* __restrict__ newpc, float* __restrict__ out) {
  const int t = blockIdx.x * blockDim.x + threadIdx.x;
  if (t >= B_ * 3 * M_) return;
  const int m = t % M_, d = (t / M_) % 3, b = t / (3 * M_);
  const float v = pc[((size_t)b * 3 + d) * N_ + samp[b * M_ + m]];
  newpc[((size_t)b * 3 + d) * M_ + m] = v;
  out[((size_t)b * 3 + d) * M_ + m]   = v;
}

// ----------------------------------------------------------------------
// Ball query: one wave32 per center. Emits first-K in-radius indices in
// ascending order (matches reference top_k(-cand) semantics), pads with
// the first found index.
// ----------------------------------------------------------------------
__global__ __launch_bounds__(256) void ballq_kernel(const float* __restrict__ pc,
                                                    const float* __restrict__ newpc,
                                                    int* __restrict__ gidx,
                                                    float r2, int K) {
  const int gt = blockIdx.x * 256 + threadIdx.x;
  const int w = gt >> 5, lane = gt & 31;
  if (w >= B_ * M_) return;
  const int b = w >> 10, m = w & (M_ - 1);
  const float cx = newpc[((size_t)b * 3 + 0) * M_ + m];
  const float cy = newpc[((size_t)b * 3 + 1) * M_ + m];
  const float cz = newpc[((size_t)b * 3 + 2) * M_ + m];
  const float* p = pc + (size_t)b * 3 * N_;
  int* g = gidx + ((size_t)b * M_ + m) * 64;
  int count = 0, first = 0;
  for (int base = 0; base < N_ && count < K; base += 32) {
    const int j = base + lane;
    const float dx = p[j] - cx, dy = p[N_ + j] - cy, dz = p[2 * N_ + j] - cz;
    const float d2 = dx * dx + dy * dy + dz * dz;
    const bool inb = d2 < r2;
    const unsigned mask = (unsigned)__ballot(inb);
    if (count == 0 && mask) first = base + (__ffs((int)mask) - 1);
    const int pos = count + __popc(mask & ((1u << lane) - 1u));
    if (inb && pos < K) g[pos] = j;
    count += __popc(mask);
  }
  if (count > K) count = K;
  for (int s = count + lane; s < K; s += 32) g[s] = first;
}

// ----------------------------------------------------------------------
// Build X0 (f16): layout [b][col = m*K + k][c], c in [0,96):
//  c<3: xyz offsets, 3..66: grouped features, 67..95: zero pad.
// ----------------------------------------------------------------------
__global__ void buildX_kernel(const float* __restrict__ pc, const float* __restrict__ feat,
                              const float* __restrict__ newpc, const int* __restrict__ gidx,
                              _Float16* __restrict__ X, int K, int CB) {
  const size_t t = (size_t)blockIdx.x * blockDim.x + threadIdx.x;
  const size_t total = (size_t)B_ * CB * 96;
  if (t >= total) return;
  const int c = (int)(t % 96);
  const size_t col = (t / 96) % CB;
  const int b = (int)(t / ((size_t)96 * CB));
  float v = 0.f;
  if (c < 67) {
    const int m = (int)(col / K), kk = (int)(col % K);
    const int idx = gidx[((size_t)b * M_ + m) * 64 + kk];
    if (c < 3)
      v = pc[((size_t)b * 3 + c) * N_ + idx] - newpc[((size_t)b * 3 + c) * M_ + m];
    else
      v = feat[((size_t)b * 64 + (c - 3)) * N_ + idx];
  }
  X[t] = (_Float16)v;
}

// ----------------------------------------------------------------------
// Convert/pad weights to f16: Wh[COUT][CINP], zeros for c >= CIN.
// ----------------------------------------------------------------------
__global__ void convW_kernel(const float* __restrict__ W, _Float16* __restrict__ Wh,
                             int COUT, int CIN, int CINP) {
  const int t = blockIdx.x * blockDim.x + threadIdx.x;
  if (t >= COUT * CINP) return;
  const int c = t % CINP, co = t / CINP;
  Wh[t] = (c < CIN) ? (_Float16)W[co * CIN + c] : (_Float16)0.f;
}

// ----------------------------------------------------------------------
// WMMA GEMM: Y[b][col][co] = W[co][:] . X[b][col][:] + bias[co]
// Templated on (COUT, CINP) -> NT = COUT/16 tiles, NK = CINP/32 k-steps,
// all fully unrolled: per wave, per 16-col group, a branch-free sequence
// of 1 global v16h B load + NT WMMAs per k-step (A fragments from LDS,
// hoisted to VGPRs by the compiler across the col-group loop).
// Epilogue: f16 convert + b128 store only (stats handled separately).
// Fragment layouts used here (contiguous k, same order for A and B):
//   A 16x32 f16: lane L row L&15, halves k = (L>=16?16:0)+[0..15].
//   B 32x16 f16: lane L col L&15, halves k = (L>=16?16:0)+[0..15].
//   D 16x16 f32: lane L col L&15, rows (L>=16?8:0)+[0..7].
// ----------------------------------------------------------------------
template <int COUT, int CINP>
__global__ __launch_bounds__(256) void gemm_kernel(
    const _Float16* __restrict__ X, const _Float16* __restrict__ Wt,
    const float* __restrict__ bias, _Float16* __restrict__ Y, int CB) {
  constexpr int NT = COUT / 16;   // cout tiles per wave (<= 8)
  constexpr int NK = CINP / 32;   // k-steps (2 or 3)
  __shared__ _Float16 Wsh[COUT * CINP];
  const int tid = threadIdx.x, lane = tid & 31, wv = tid >> 5;
  const int b = blockIdx.y;
  {  // stage all COUT rows of W into LDS (row-major, contiguous)
    const unsigned long long* src = (const unsigned long long*)Wt;
    unsigned long long* dst = (unsigned long long*)Wsh;
    constexpr int n = (COUT * CINP) >> 2;
    for (int i = tid; i < n; i += 256) dst[i] = src[i];
  }
  __syncthreads();
  const int hs = lane >> 4;     // half-select: 0 for lanes 0-15, 1 for 16-31
  const int lc = lane & 15;
  v8f binit[NT];                // bias, preloaded once per block
#pragma unroll
  for (int t = 0; t < NT; ++t) {
    const int r0 = t * 16 + hs * 8;
#pragma unroll
    for (int r = 0; r < 8; ++r) binit[t][r] = bias[r0 + r];
  }
  const int ncg = CB >> 7;      // 128 cols per block iteration (8 waves x 16)
  for (int cg = blockIdx.x; cg < ncg; cg += gridDim.x) {
    const int col = (cg << 7) + (wv << 4) + lc;
    const _Float16* Xp = X + ((size_t)b * CB + col) * CINP + hs * 16;
    v8f acc[NT];
#pragma unroll
    for (int t = 0; t < NT; ++t) acc[t] = binit[t];
#pragma unroll
    for (int kb = 0; kb < NK; ++kb) {
      const int k0 = kb * 32;
      const v16h bf = *(const v16h*)(Xp + k0);  // 32B contiguous, aligned
#pragma unroll
      for (int t = 0; t < NT; ++t) {
        const _Float16* Wp = Wsh + (t * 16 + lc) * CINP + k0 + hs * 16;
        const v8h a0 = *(const v8h*)(Wp);
        const v8h a1 = *(const v8h*)(Wp + 8);
        v16h af;
#pragma unroll
        for (int q = 0; q < 8; ++q) { af[q] = a0[q]; af[8 + q] = a1[q]; }
        acc[t] = __builtin_amdgcn_wmma_f32_16x16x32_f16(
            false, af, false, bf, (short)0, acc[t], false, false);
      }
    }
    // Lean epilogue: convert to f16, one b128 store per tile.
#pragma unroll
    for (int t = 0; t < NT; ++t) {
      const int r0 = t * 16 + hs * 8;
      _Float16* Yp = Y + ((size_t)b * CB + col) * COUT + r0;
      v8h yh;
#pragma unroll
      for (int r = 0; r < 8; ++r) yh[r] = (_Float16)acc[t][r];
      *(v8h*)Yp = yh;
    }
  }
}

// ----------------------------------------------------------------------
// Per-channel sum / sumsq over all columns of Y (bandwidth bound).
// Thread owns a fixed channel octet; v8h loads; register accumulation in
// fixed column order; fixed-order LDS tree -> per-block partials.
// Fully deterministic. Requires NC % (SGX_*32) == 0 (holds: 512K / 1M).
// ----------------------------------------------------------------------
template <int COUT>
__global__ __launch_bounds__((COUT / 8) * 32) void stats_kernel(
    const _Float16* __restrict__ Y, float* __restrict__ partials, int NC) {
  constexpr int NQ = COUT / 8;   // channel octets per column
  constexpr int CPB = 32;        // column slots per block
  __shared__ float lsum[CPB * COUT];
  __shared__ float lsq[CPB * COUT];
  const int tid = threadIdx.x;
  const int q = tid % NQ;        // octet index
  const int s = tid / NQ;        // column slot
  float sum[8], sq[8];
#pragma unroll
  for (int r = 0; r < 8; ++r) { sum[r] = 0.f; sq[r] = 0.f; }
  for (int col = blockIdx.x * CPB + s; col < NC; col += SGX_ * CPB) {
    const v8h y = *(const v8h*)(Y + (size_t)col * COUT + q * 8);
#pragma unroll
    for (int r = 0; r < 8; ++r) {
      const float v = (float)y[r];
      sum[r] += v; sq[r] += v * v;
    }
  }
#pragma unroll
  for (int r = 0; r < 8; ++r) {
    lsum[s * COUT + q * 8 + r] = sum[r];
    lsq [s * COUT + q * 8 + r] = sq[r];
  }
  __syncthreads();
  if (tid < COUT) {
    float ts = 0.f, tq = 0.f;
#pragma unroll
    for (int k = 0; k < CPB; ++k) { ts += lsum[k * COUT + tid]; tq += lsq[k * COUT + tid]; }
    float* pp = partials + (size_t)blockIdx.x * (2 * COUT);
    pp[2 * tid] = ts; pp[2 * tid + 1] = tq;
  }
}

// ----------------------------------------------------------------------
// Finalize BN: fixed-order partial sum -> per-channel scale/shift.
// partials layout: [SGX_][COUT*2].
// ----------------------------------------------------------------------
__global__ void bn_finalize(const float* __restrict__ partials,
                            const float* __restrict__ g, const float* __restrict__ be,
                            float* __restrict__ st, int COUT, float invcnt) {
  const int ch = blockIdx.x * blockDim.x + threadIdx.x;
  if (ch >= COUT) return;
  float s = 0.f, q = 0.f;
  for (int blk = 0; blk < SGX_; ++blk) {
    const float* pp = partials + (size_t)blk * (2 * COUT) + ch * 2;
    s += pp[0]; q += pp[1];
  }
  const float mu  = s * invcnt;
  const float var = q * invcnt - mu * mu;
  const float sc  = g[ch] * rsqrtf(var + 1e-5f);
  st[ch * 2]     = sc;
  st[ch * 2 + 1] = be[ch] - mu * sc;
}

// ----------------------------------------------------------------------
// In-place normalize + ReLU (8 channels / thread).
// ----------------------------------------------------------------------
__global__ void bn_relu_vec8(_Float16* __restrict__ Y, const float* __restrict__ st,
                             int COUT, size_t nvec) {
  const size_t t = (size_t)blockIdx.x * blockDim.x + threadIdx.x;
  if (t >= nvec) return;
  const size_t base = t * 8;
  const int co0 = (int)(base % COUT);
  v8h y = *(v8h*)(Y + base);
  v8h o;
#pragma unroll
  for (int r = 0; r < 8; ++r) {
    const float v = (float)y[r] * st[(co0 + r) * 2] + st[(co0 + r) * 2 + 1];
    o[r] = (_Float16)fmaxf(v, 0.f);
  }
  *(v8h*)(Y + base) = o;
}

// ----------------------------------------------------------------------
// Last layer: normalize + ReLU + max over K -> d_out (f32). COUT = 128.
// ----------------------------------------------------------------------
__global__ __launch_bounds__(128) void bn_relu_maxk(const _Float16* __restrict__ Y,
                                                    const float* __restrict__ st,
                                                    float* __restrict__ out2,
                                                    int K, int choff) {
  const int co = threadIdx.x;
  const int m = blockIdx.x, b = blockIdx.y;
  const _Float16* yp = Y + (((size_t)b * M_ + m) * K) * 128 + co;
  const float sc = st[co * 2], sh = st[co * 2 + 1];
  float mx = -1e30f;
  for (int kk = 0; kk < K; ++kk) {
    float v = (float)yp[(size_t)kk * 128] * sc + sh;
    v = fmaxf(v, 0.f);
    mx = fmaxf(mx, v);
  }
  out2[((size_t)b * 256 + choff + co) * M_ + m] = mx;
}

// =====================================================================
// Host orchestration
// =====================================================================
extern "C" void kernel_launch(void* const* d_in, const int* in_sizes, int n_in,
                              void* d_out, int out_size, void* d_ws, size_t ws_size,
                              hipStream_t stream) {
  (void)in_sizes; (void)n_in; (void)out_size; (void)ws_size;
  const float* pc   = (const float*)d_in[0];
  const float* feat = (const float*)d_in[1];
  auto IN = [&](int i, int j, int p) { return (const float*)d_in[2 + (i * 3 + j) * 4 + p]; };

  // workspace carve (256B aligned)
  char* ws = (char*)d_ws;
  size_t off = 0;
  auto carve = [&](size_t bytes) { char* r = ws + off; off = (off + bytes + 255) & ~(size_t)255; return r; };
  int*      samp     = (int*)carve((size_t)B_ * M_ * 4);
  float*    newpc    = (float*)carve((size_t)B_ * 3 * M_ * 4);
  int*      gidx     = (int*)carve((size_t)B_ * M_ * 64 * 4);
  _Float16* Wh       = (_Float16*)carve((size_t)128 * 96 * 2);
  float*    partials = (float*)carve((size_t)SGX_ * 256 * 4);
  float*    st       = (float*)carve((size_t)128 * 2 * 4);
  _Float16* bufA     = (_Float16*)carve((size_t)B_ * 65536 * 96 * 2);   // 201 MB
  _Float16* bufB     = (_Float16*)carve((size_t)B_ * 65536 * 128 * 2);  // 268 MB

  float* out  = (float*)d_out;
  float* out2 = out + (size_t)B_ * 3 * M_;

  fps_kernel<<<B_, 1024, 0, stream>>>(pc, samp);
  gather_newpc<<<(B_ * 3 * M_ + 255) / 256, 256, 0, stream>>>(pc, samp, newpc, out);

  const int   KS_[2]     = {32, 64};
  const float R2_[2]     = {(float)(0.1 * 0.1), (float)(0.2 * 0.2)};
  const int   SPEC[2][3] = {{64, 64, 128}, {64, 96, 128}};

  auto launch_gemm = [&](int COUT, int CINP, const _Float16* Xc, const float* bias,
                         _Float16* Yb, int CB) {
    const dim3 grid(GX_, B_);
    if      (COUT == 64  && CINP == 96)
      gemm_kernel<64, 96><<<grid, 256, 0, stream>>>(Xc, Wh, bias, Yb, CB);
    else if (COUT == 64  && CINP == 64)
      gemm_kernel<64, 64><<<grid, 256, 0, stream>>>(Xc, Wh, bias, Yb, CB);
    else if (COUT == 96  && CINP == 64)
      gemm_kernel<96, 64><<<grid, 256, 0, stream>>>(Xc, Wh, bias, Yb, CB);
    else if (COUT == 128 && CINP == 64)
      gemm_kernel<128, 64><<<grid, 256, 0, stream>>>(Xc, Wh, bias, Yb, CB);
    else if (COUT == 128 && CINP == 96)
      gemm_kernel<128, 96><<<grid, 256, 0, stream>>>(Xc, Wh, bias, Yb, CB);
  };

  auto launch_stats = [&](int COUT, const _Float16* Yb, int NC) {
    if      (COUT == 64)  stats_kernel<64><<<SGX_, 256, 0, stream>>>(Yb, partials, NC);
    else if (COUT == 96)  stats_kernel<96><<<SGX_, 384, 0, stream>>>(Yb, partials, NC);
    else if (COUT == 128) stats_kernel<128><<<SGX_, 512, 0, stream>>>(Yb, partials, NC);
  };

  for (int i = 0; i < 2; ++i) {
    const int K = KS_[i];
    const int CB = M_ * K;
    ballq_kernel<<<(B_ * M_ * 32 + 255) / 256, 256, 0, stream>>>(pc, newpc, gidx, R2_[i], K);
    {
      const size_t total = (size_t)B_ * CB * 96;
      buildX_kernel<<<(unsigned)((total + 255) / 256), 256, 0, stream>>>(
          pc, feat, newpc, gidx, bufA, K, CB);
    }
    _Float16* Xc = bufA;
    int CINP = 96;
    for (int j = 0; j < 3; ++j) {
      const int COUT = SPEC[i][j];
      const int CIN  = (j == 0) ? 67 : SPEC[i][j - 1];
      convW_kernel<<<(COUT * CINP + 255) / 256, 256, 0, stream>>>(IN(i, j, 0), Wh, COUT, CIN, CINP);
      _Float16* Yb = (j & 1) ? bufA : bufB;  // j0->B, j1->A, j2->B (ping-pong)
      launch_gemm(COUT, CINP, Xc, IN(i, j, 1), Yb, CB);
      launch_stats(COUT, Yb, B_ * CB);
      bn_finalize<<<1, 128, 0, stream>>>(partials, IN(i, j, 2), IN(i, j, 3), st,
                                         COUT, 1.0f / (float)((size_t)B_ * CB));
      if (j < 2) {
        const size_t nv = (size_t)B_ * CB * COUT / 8;
        bn_relu_vec8<<<(unsigned)((nv + 255) / 256), 256, 0, stream>>>(Yb, st, COUT, nv);
        Xc = Yb;
        CINP = COUT;  // 64 or 96, both multiples of 32
      } else {
        bn_relu_maxk<<<dim3(M_, B_), 128, 0, stream>>>(Yb, st, out2, K, 128 * i);
      }
    }
  }
}
